// VariableSelectionNetwork_14611478741005
// MI455X (gfx1250) — matrix operand order repaired and checked by reading, not compiled, not run on hardware
//
#include <hip/hip_runtime.h>
#include <hip/hip_bf16.h>

// ---------------------------------------------------------------------------
// Variable Selection Network for MI455X (gfx1250, wave32, WMMA).
// Compute-bound (17.2 GFLOP vs ~15MB traffic) -> bf16 WMMA w/ f32 accumulate.
// Round 2: 32-row M-tiles in the main kernel (2x B-fragment reuse, half the
// barrier/LN overhead per FLOP) + global_prefetch of next feature's weights.
// ---------------------------------------------------------------------------

typedef __attribute__((ext_vector_type(16))) __bf16 v16bf;
typedef __attribute__((ext_vector_type(8)))  __bf16 v8bf;
typedef __attribute__((ext_vector_type(8)))  float  v8f;

#define BT   8192
#define NFEA 32
#define DM   256
#define HH   64

__device__ __forceinline__ unsigned short f2bf(float f) {
    unsigned int u = __float_as_uint(f);
    return (unsigned short)((u + 0x7FFFu + ((u >> 16) & 1u)) >> 16);
}

// A fragment (16-bit, 16x32): lane holds K = hl*16+{0..7} then hl*16+8+{0..7}
__device__ __forceinline__ v16bf frag_lds(const unsigned short* p0,
                                          const unsigned short* p1) {
    v8bf lo = *(const v8bf*)p0;
    v8bf hi = *(const v8bf*)p1;
    return __builtin_shufflevector(lo, hi, 0,1,2,3,4,5,6,7,8,9,10,11,12,13,14,15);
}

__device__ __forceinline__ v8f wmma_bf16(v16bf a, v16bf b, v8f c) {
    return __builtin_amdgcn_wmma_f32_16x16x32_bf16(false, a, false, b,
                                                   (short)0, c, false, false);
}

__device__ __forceinline__ float elu_f(float v) {
    return v > 0.f ? v : __expf(v) - 1.f;
}
__device__ __forceinline__ float sigm_f(float v) {
    return 1.f / (1.f + __expf(-v));
}

// ---------------------------------------------------------------------------
// Kernel 1: fp32 -> bf16 weight repack (layouts already B-fragment friendly)
// ---------------------------------------------------------------------------
__global__ __launch_bounds__(256) void vsn_convert(
    const float* g1, const float* gg, const float* gf2,
    const float* fg, const float* ff,
    unsigned short* o_g1, unsigned short* o_gg, unsigned short* o_gf2,
    unsigned short* o_fg, unsigned short* o_ff) {
    int i = blockIdx.x * 256 + threadIdx.x;
    if (i < 8192) {
        o_g1[i]  = f2bf(g1[i]);
        o_gg[i]  = f2bf(gg[i]);
        o_gf2[i] = f2bf(gf2[i]);
    }
    if (i < NFEA * DM * HH) {       // 524288, grid sized exactly
        o_fg[i] = f2bf(fg[i]);
        o_ff[i] = f2bf(ff[i]);
    }
}

// ---------------------------------------------------------------------------
// Kernel 2: selection GRN -> softmax weights [BT, 32] (4 waves / 16 rows)
// ---------------------------------------------------------------------------
__global__ __launch_bounds__(128) void vsn_sel(
    const float* __restrict__ x,
    const unsigned short* __restrict__ Wg1, const float* __restrict__ b1,
    const unsigned short* __restrict__ Wgg, const float* __restrict__ bgg,
    const unsigned short* __restrict__ Wg2, const float* __restrict__ bg2,
    const float* __restrict__ lng, const float* __restrict__ lnb,
    float* __restrict__ wsel) {
    __shared__ float xt[16 * 32];
    __shared__ __align__(32) unsigned short xb[16 * 40];   // padded stride 40
    __shared__ __align__(32) unsigned short hb[16 * 264];  // padded stride 264
    __shared__ float ab[16 * 32];
    __shared__ float h2b[16 * 32];

    const int tid = threadIdx.x;
    const int r0  = blockIdx.x * 16;
    for (int i = tid; i < 512; i += 128) {
        int r = i >> 5, cc = i & 31;
        float v = x[(r0 + r) * 32 + cc];
        xt[r * 32 + cc] = v;
        xb[r * 40 + cc] = f2bf(v);
    }
    __syncthreads();

    const int w = tid >> 5, lane = tid & 31;
    const int c = lane & 15, hl = lane >> 4;

    // GEMM1: h = elu(x @ g_fc1_w^T + b1)  -> [16 x 256] bf16 in LDS
    #pragma unroll
    for (int tt = 0; tt < 4; ++tt) {
        int ct = w * 4 + tt;
        v16bf a = frag_lds(&xb[c * 40 + hl * 16], &xb[c * 40 + hl * 16 + 8]);
        int d = ct * 16 + c;
        v16bf b = *(const v16bf*)&Wg1[d * 32 + hl * 16];
        v8f acc = {};
        acc = wmma_bf16(a, b, acc);
        float bias = b1[d];
        #pragma unroll
        for (int j = 0; j < 8; ++j) {
            int r = j + 8 * hl;
            hb[r * 264 + d] = f2bf(elu_f(acc[j] + bias));
        }
    }
    __syncthreads();

    // GEMM2/3: waves 0,1 -> gate; waves 2,3 -> fc2;  [16 x 256] @ W^T -> [16 x 32]
    {
        const unsigned short* W = (w < 2) ? Wgg : Wg2;
        int ct = w & 1;
        int f  = ct * 16 + c;
        v8f acc = {};
        #pragma unroll
        for (int ks = 0; ks < 8; ++ks) {
            v16bf a = frag_lds(&hb[c * 264 + ks * 32 + hl * 16],
                               &hb[c * 264 + ks * 32 + hl * 16 + 8]);
            v16bf b = *(const v16bf*)&W[f * 256 + ks * 32 + hl * 16];
            acc = wmma_bf16(a, b, acc);
        }
        float bias = (w < 2) ? bgg[f] : bg2[f];
        float* dst = (w < 2) ? ab : h2b;
        #pragma unroll
        for (int j = 0; j < 8; ++j)
            dst[(j + 8 * hl) * 32 + f] = acc[j] + bias;
    }
    __syncthreads();

    // LN + softmax per row (wave handles 4 rows; lane = feature)
    for (int rr = 0; rr < 4; ++rr) {
        int r = w * 4 + rr;
        int f = lane;
        float y = sigm_f(ab[r * 32 + f]) * h2b[r * 32 + f] + xt[r * 32 + f];
        float s = y;
        #pragma unroll
        for (int m = 1; m < 32; m <<= 1) s += __shfl_xor(s, m, 32);
        float mean = s * (1.f / 32.f);
        float dv = (y - mean) * (y - mean);
        #pragma unroll
        for (int m = 1; m < 32; m <<= 1) dv += __shfl_xor(dv, m, 32);
        float rstd = rsqrtf(dv * (1.f / 32.f) + 1e-5f);
        float yn = (y - mean) * rstd * lng[f] + lnb[f];
        float mx = yn;
        #pragma unroll
        for (int m = 1; m < 32; m <<= 1) mx = fmaxf(mx, __shfl_xor(mx, m, 32));
        float e = __expf(yn - mx);
        float se = e;
        #pragma unroll
        for (int m = 1; m < 32; m <<= 1) se += __shfl_xor(se, m, 32);
        wsel[(r0 + r) * 32 + f] = e / se;
    }
}

// ---------------------------------------------------------------------------
// Kernel 3: per-feature GRNs + weighted combine.
// 8 waves / block, 32 rows (2 M-tiles) / block: each B fragment feeds 2 WMMAs.
// ---------------------------------------------------------------------------
__global__ __launch_bounds__(256) void vsn_feat(
    const float* __restrict__ x, const float* __restrict__ wsel,
    const float* __restrict__ f1w, const float* __restrict__ f1b,
    const unsigned short* __restrict__ Wg, const float* __restrict__ gbias,
    const unsigned short* __restrict__ Wf, const float* __restrict__ fbias,
    const float* __restrict__ skw, const float* __restrict__ skb,
    const float* __restrict__ lng, const float* __restrict__ lnb,
    float* __restrict__ out) {
    __shared__ float xt[32 * 32];
    __shared__ float wt[32 * 32];
    __shared__ __align__(32) unsigned short hf[32 * 72];  // padded stride 72
    __shared__ float w1[NFEA * HH];
    __shared__ float bb1[NFEA * HH];
    __shared__ float redS[32 * 8];
    __shared__ float redQ[32 * 8];
    __shared__ float rowM[32];
    __shared__ float rowR[32];

    const int tid = threadIdx.x;
    const int r0  = blockIdx.x * 32;
    for (int i = tid; i < 1024; i += 256) {
        int r = i >> 5, cc = i & 31;
        xt[i] = x[(r0 + r) * 32 + cc];
        wt[i] = wsel[(r0 + r) * 32 + cc];
    }
    for (int i = tid; i < NFEA * HH; i += 256) { w1[i] = f1w[i]; bb1[i] = f1b[i]; }

    const int w = tid >> 5, lane = tid & 31;
    const int c = lane & 15, hl = lane >> 4;
    const int d0 = (2 * w) * 16 + c;
    const int d1 = (2 * w + 1) * 16 + c;

    v8f selA0 = {}, selA1 = {}, selB0 = {}, selB1 = {};
    __syncthreads();

    for (int n = 0; n < NFEA; ++n) {
        // hf = elu(x[:,n] * w1[n,:] + b1[n,:])  -> bf16 LDS [32 x 64]
        for (int i = tid; i < 2048; i += 256) {
            int r = i >> 6, h = i & 63;
            float v = xt[r * 32 + n] * w1[n * HH + h] + bb1[n * HH + h];
            hf[r * 72 + h] = f2bf(elu_f(v));
        }
        __syncthreads();

        // two GEMMs [32,64]x[64,256]; wave owns cols 32w..32w+31, both M-tiles
        const unsigned short* Wgn = Wg + n * (DM * HH);
        const unsigned short* Wfn = Wf + n * (DM * HH);
        v8f cgA0 = {}, cgA1 = {}, cfA0 = {}, cfA1 = {};
        v8f cgB0 = {}, cgB1 = {}, cfB0 = {}, cfB1 = {};
        #pragma unroll
        for (int ks = 0; ks < 2; ++ks) {
            v16bf aA = frag_lds(&hf[c * 72 + ks * 32 + hl * 16],
                                &hf[c * 72 + ks * 32 + hl * 16 + 8]);
            v16bf aB = frag_lds(&hf[(16 + c) * 72 + ks * 32 + hl * 16],
                                &hf[(16 + c) * 72 + ks * 32 + hl * 16 + 8]);
            v16bf bg0 = *(const v16bf*)&Wgn[d0 * HH + ks * 32 + hl * 16];
            v16bf bg1 = *(const v16bf*)&Wgn[d1 * HH + ks * 32 + hl * 16];
            v16bf bf0 = *(const v16bf*)&Wfn[d0 * HH + ks * 32 + hl * 16];
            v16bf bf1 = *(const v16bf*)&Wfn[d1 * HH + ks * 32 + hl * 16];
            cgA0 = wmma_bf16(aA, bg0, cgA0);
            cgB0 = wmma_bf16(aB, bg0, cgB0);
            cgA1 = wmma_bf16(aA, bg1, cgA1);
            cgB1 = wmma_bf16(aB, bg1, cgB1);
            cfA0 = wmma_bf16(aA, bf0, cfA0);
            cfB0 = wmma_bf16(aB, bf0, cfB0);
            cfA1 = wmma_bf16(aA, bf1, cfA1);
            cfB1 = wmma_bf16(aB, bf1, cfB1);
        }

        // prefetch next feature's weight rows into cache (global_prefetch_b8)
        {
            int n2 = (n + 1 < NFEA) ? n + 1 : n;
            const unsigned short* Pg = Wg + n2 * (DM * HH);
            const unsigned short* Pf = Wf + n2 * (DM * HH);
            __builtin_prefetch(&Pg[d0 * HH], 0, 0);
            __builtin_prefetch(&Pg[d1 * HH], 0, 0);
            __builtin_prefetch(&Pf[d0 * HH], 0, 0);
            __builtin_prefetch(&Pf[d1 * HH], 0, 0);
        }

        // fused gate * fc2 + skip, produce y; deterministic row sums
        float gb0 = gbias[n * DM + d0], gb1 = gbias[n * DM + d1];
        float fb0 = fbias[n * DM + d0], fb1 = fbias[n * DM + d1];
        float sw0 = skw[n * DM + d0],   sw1 = skw[n * DM + d1];
        float sb0 = skb[n * DM + d0],   sb1 = skb[n * DM + d1];
        #pragma unroll
        for (int j = 0; j < 8; ++j) {
            int rA = j + 8 * hl;
            int rB = 16 + rA;
            float xvA = xt[rA * 32 + n];
            float xvB = xt[rB * 32 + n];
            float yA0 = sigm_f(cgA0[j] + gb0) * (cfA0[j] + fb0) + xvA * sw0 + sb0;
            float yA1 = sigm_f(cgA1[j] + gb1) * (cfA1[j] + fb1) + xvA * sw1 + sb1;
            float yB0 = sigm_f(cgB0[j] + gb0) * (cfB0[j] + fb0) + xvB * sw0 + sb0;
            float yB1 = sigm_f(cgB1[j] + gb1) * (cfB1[j] + fb1) + xvB * sw1 + sb1;
            cgA0[j] = yA0; cgA1[j] = yA1; cgB0[j] = yB0; cgB1[j] = yB1;
            float sA = yA0 + yA1, qA = yA0 * yA0 + yA1 * yA1;
            float sB = yB0 + yB1, qB = yB0 * yB0 + yB1 * yB1;
            sA += __shfl_xor(sA, 1, 32); sA += __shfl_xor(sA, 2, 32);
            sA += __shfl_xor(sA, 4, 32); sA += __shfl_xor(sA, 8, 32);
            qA += __shfl_xor(qA, 1, 32); qA += __shfl_xor(qA, 2, 32);
            qA += __shfl_xor(qA, 4, 32); qA += __shfl_xor(qA, 8, 32);
            sB += __shfl_xor(sB, 1, 32); sB += __shfl_xor(sB, 2, 32);
            sB += __shfl_xor(sB, 4, 32); sB += __shfl_xor(sB, 8, 32);
            qB += __shfl_xor(qB, 1, 32); qB += __shfl_xor(qB, 2, 32);
            qB += __shfl_xor(qB, 4, 32); qB += __shfl_xor(qB, 8, 32);
            if (c == 0) {
                redS[rA * 8 + w] = sA; redQ[rA * 8 + w] = qA;
                redS[rB * 8 + w] = sB; redQ[rB * 8 + w] = qB;
            }
        }
        __syncthreads();
        if (tid < 32) {
            float s = 0.f, q = 0.f;
            #pragma unroll
            for (int k = 0; k < 8; ++k) { s += redS[tid * 8 + k]; q += redQ[tid * 8 + k]; }
            float m = s * (1.f / 256.f);
            rowM[tid] = m;
            rowR[tid] = rsqrtf(q * (1.f / 256.f) - m * m + 1e-5f);
        }
        __syncthreads();

        // LayerNorm + weighted accumulation into persistent accumulators
        float g0 = lng[n * DM + d0], g1v = lng[n * DM + d1];
        float l0 = lnb[n * DM + d0], l1  = lnb[n * DM + d1];
        #pragma unroll
        for (int j = 0; j < 8; ++j) {
            int rA = j + 8 * hl;
            int rB = 16 + rA;
            float mA = rowM[rA], rsA = rowR[rA];
            float mB = rowM[rB], rsB = rowR[rB];
            float wwA = wt[rA * 32 + n];
            float wwB = wt[rB * 32 + n];
            selA0[j] += wwA * ((cgA0[j] - mA) * rsA * g0 + l0);
            selA1[j] += wwA * ((cgA1[j] - mA) * rsA * g1v + l1);
            selB0[j] += wwB * ((cgB0[j] - mB) * rsB * g0 + l0);
            selB1[j] += wwB * ((cgB1[j] - mB) * rsB * g1v + l1);
        }
        // no trailing barrier needed: next-iteration hf/redS writes are fenced
        // by the two barriers above before any re-read.
    }

    #pragma unroll
    for (int j = 0; j < 8; ++j) {
        int rA = r0 + j + 8 * hl;
        int rB = rA + 16;
        out[rA * DM + d0] = selA0[j];
        out[rA * DM + d1] = selA1[j];
        out[rB * DM + d0] = selB0[j];
        out[rB * DM + d1] = selB1[j];
    }
}

// ---------------------------------------------------------------------------
// Kernel 4: feat_weights = mean over T of wsel (fixed-order, deterministic)
// ---------------------------------------------------------------------------
__global__ __launch_bounds__(512) void vsn_fw(const float* __restrict__ wsel,
                                              float* __restrict__ fw) {
    int i = threadIdx.x;                  // 512 = 16 batches * 32 features
    int b = i >> 5, f = i & 31;
    float s = 0.f;
    for (int t = 0; t < 512; ++t) s += wsel[(b * 512 + t) * 32 + f];
    fw[i] = s * (1.f / 512.f);
}

// ---------------------------------------------------------------------------
extern "C" void kernel_launch(void* const* d_in, const int* in_sizes, int n_in,
                              void* d_out, int out_size, void* d_ws, size_t ws_size,
                              hipStream_t stream) {
    const float* x       = (const float*)d_in[0];
    const float* g_fc1_w = (const float*)d_in[1];
    const float* g_fc1_b = (const float*)d_in[2];
    const float* g_gate_w= (const float*)d_in[3];
    const float* g_gate_b= (const float*)d_in[4];
    const float* g_fc2_w = (const float*)d_in[5];
    const float* g_fc2_b = (const float*)d_in[6];
    const float* g_ln_g  = (const float*)d_in[7];
    const float* g_ln_b  = (const float*)d_in[8];
    const float* f_fc1_w = (const float*)d_in[9];
    const float* f_fc1_b = (const float*)d_in[10];
    const float* f_gate_w= (const float*)d_in[11];
    const float* f_gate_b= (const float*)d_in[12];
    const float* f_fc2_w = (const float*)d_in[13];
    const float* f_fc2_b = (const float*)d_in[14];
    const float* f_skip_w= (const float*)d_in[15];
    const float* f_skip_b= (const float*)d_in[16];
    const float* f_ln_g  = (const float*)d_in[17];
    const float* f_ln_b  = (const float*)d_in[18];

    char* ws = (char*)d_ws;
    float*          wsel = (float*)(ws);                               // 1 MiB
    unsigned short* G1   = (unsigned short*)(ws + 1048576);            // 16 KiB
    unsigned short* GG   = (unsigned short*)(ws + 1048576 + 16384);    // 16 KiB
    unsigned short* GF2  = (unsigned short*)(ws + 1048576 + 32768);    // 16 KiB
    unsigned short* FG   = (unsigned short*)(ws + 1048576 + 49152);    // 1 MiB
    unsigned short* FF   = (unsigned short*)(ws + 2097152 + 49152);    // 1 MiB

    vsn_convert<<<(NFEA * DM * HH) / 256, 256, 0, stream>>>(
        g_fc1_w, g_gate_w, g_fc2_w, f_gate_w, f_fc2_w, G1, GG, GF2, FG, FF);

    vsn_sel<<<BT / 16, 128, 0, stream>>>(
        x, G1, g_fc1_b, GG, g_gate_b, GF2, g_fc2_b, g_ln_g, g_ln_b, wsel);

    vsn_feat<<<BT / 32, 256, 0, stream>>>(
        x, wsel, f_fc1_w, f_fc1_b, FG, f_gate_b, FF, f_fc2_b,
        f_skip_w, f_skip_b, f_ln_g, f_ln_b, (float*)d_out);

    vsn_fw<<<1, 512, 0, stream>>>(wsel, (float*)d_out + BT * DM);
}
